// ViTChess_28561532518665
// MI455X (gfx1250) — matrix operand order repaired and checked by reading, not compiled, not run on hardware
//
#include <hip/hip_runtime.h>
#include <hip/hip_bf16.h>
#include <stdint.h>

typedef __attribute__((ext_vector_type(16))) _Float16 v16h;
typedef __attribute__((ext_vector_type(8)))  _Float16 v8h;
typedef __attribute__((ext_vector_type(8)))  float    v8f;

#define CB   768
#define HB   12
#define DHB  64
#define LATB 256
#define NTOK 65
#define BB   512

// ---------------------------------------------------------------------------
// WMMA GEMM, async-copy edition:
//   C[M,N] (+)= A[M,K] * BT[N,K]^T      A, BT: f16 row-major; C: f32
// Block tile 128x128x64, 256 threads = 8 waves, wave tile 32x64 (2x4 WMMA).
// Tiles staged with global_load_async_to_lds_b128 (ASYNCcnt), double-buffered.
// Requires K % 64 == 0 and M % 128 == 0 (true at every call site). N guarded.
// ---------------------------------------------------------------------------
#define BM 128
#define BN 128
#define BK 64
#define KP 8
#define ROWH (BK + KP)                 // 72 halves = 144B row stride (16B mult.)

__global__ __launch_bounds__(256)
void gemm_f16wmma_async(const _Float16* __restrict__ A, int lda,
                        const _Float16* __restrict__ BT, int ldb,
                        float* __restrict__ C, int ldc,
                        int M, int N, int K, int accum)
{
    __shared__ __align__(16) _Float16 As[2][BM][ROWH];
    __shared__ __align__(16) _Float16 Bs[2][BN][ROWH];

    const int tid   = threadIdx.x;
    const int lane  = tid & 31;
    const int w     = tid >> 5;
    const int wm    = (w & 3) * 32;
    const int wn    = (w >> 2) * 64;
    const int tileM = blockIdx.y * BM;
    const int tileN = blockIdx.x * BN;
    const int l16   = lane & 15;
    const int hi    = lane >> 4;

    // async-copy coordinates: each thread moves 4 x 16B chunks for A and B
    const int crow = tid >> 3;                    // row 0..31 (+32 per pass)
    const int cch  = tid & 7;                     // 16B chunk within 128B row

    const unsigned aBase = (unsigned)(uintptr_t)&As[0][0][0];
    const unsigned bBase = (unsigned)(uintptr_t)&Bs[0][0][0];
    const unsigned tileBytes = (unsigned)(BM * ROWH * 2);

    v8f acc[2][4];
#pragma unroll
    for (int i = 0; i < 2; ++i)
#pragma unroll
        for (int j = 0; j < 4; ++j)
#pragma unroll
            for (int r = 0; r < 8; ++r) acc[i][j][r] = 0.0f;

    auto issue = [&](int k0, int buf) {
#pragma unroll
        for (int p = 0; p < 4; ++p) {             // A tile: 128 rows x 128B
            const int r = crow + p * 32;
            const unsigned lofs = aBase + (unsigned)buf * tileBytes
                                + (unsigned)(r * ROWH + cch * 8) * 2u;
            const _Float16* gp = A + (size_t)(tileM + r) * lda + (k0 + cch * 8);
            asm volatile("global_load_async_to_lds_b128 %0, %1, off"
                         :: "v"(lofs), "v"((unsigned long long)(uintptr_t)gp)
                         : "memory");
        }
#pragma unroll
        for (int p = 0; p < 4; ++p) {             // B tile: 128 rows x 128B
            const int r = crow + p * 32;
            int gn = tileN + r; gn = (gn < N) ? gn : (N - 1);   // clamp, cols >= N never stored
            const unsigned lofs = bBase + (unsigned)buf * tileBytes
                                + (unsigned)(r * ROWH + cch * 8) * 2u;
            const _Float16* gp = BT + (size_t)gn * ldb + (k0 + cch * 8);
            asm volatile("global_load_async_to_lds_b128 %0, %1, off"
                         :: "v"(lofs), "v"((unsigned long long)(uintptr_t)gp)
                         : "memory");
        }
    };

    issue(0, 0);
    asm volatile("s_wait_asynccnt 0x0" ::: "memory");
    __syncthreads();

    const int nk = K / BK;
    for (int t = 0; t < nk; ++t) {
        const int buf = t & 1;
        if (t + 1 < nk) issue((t + 1) * BK, buf ^ 1);   // overlap copy with math

        const _Float16 (*Asb)[ROWH] = As[buf];
        const _Float16 (*Bsb)[ROWH] = Bs[buf];
#pragma unroll
        for (int ks = 0; ks < BK; ks += 32) {
            union HV { v16h v; v8h h[2]; };
            v16h a[2], b[4];
#pragma unroll
            for (int i = 0; i < 2; ++i) {               // A frags (16x32)
                const int mr = wm + i * 16 + l16;
                const int kb = ks + hi * 8;
                HV u;
                u.h[0] = *(const v8h*)&Asb[mr][kb];
                u.h[1] = *(const v8h*)&Asb[mr][kb + 16];
                a[i] = u.v;
            }
#pragma unroll
            for (int j = 0; j < 4; ++j) {               // B frags (32x16)
                const int nc = wn + j * 16 + l16;
                const int kb = ks + hi * 8;
                HV u;
                u.h[0] = *(const v8h*)&Bsb[nc][kb];
                u.h[1] = *(const v8h*)&Bsb[nc][kb + 16];
                b[j] = u.v;
            }
#pragma unroll
            for (int i = 0; i < 2; ++i)
#pragma unroll
                for (int j = 0; j < 4; ++j)
                    acc[i][j] = __builtin_amdgcn_wmma_f32_16x16x32_f16(
                        false, a[i], false, b[j], (short)0, acc[i][j], false, false);
        }
        asm volatile("s_wait_asynccnt 0x0" ::: "memory");
        __syncthreads();
    }

    // store (C/D layout: VGPR r -> M = r + 8*hi, N = lane%16)
#pragma unroll
    for (int i = 0; i < 2; ++i)
#pragma unroll
        for (int j = 0; j < 4; ++j) {
            const int gn = tileN + wn + j * 16 + l16;
            if (gn >= N) continue;
#pragma unroll
            for (int r = 0; r < 8; ++r) {
                const int gm = tileM + wm + i * 16 + hi * 8 + r;
                if (gm >= M) continue;
                const size_t idx = (size_t)gm * ldc + gn;
                const float v = acc[i][j][r];
                if (accum) C[idx] += v; else C[idx] = v;
            }
        }
}

// ---------------------------------------------------------------------------
// Weight convert + transpose: W[K][N] f32 -> BT[N][K] f16 (LDS-tiled, coalesced)
// ---------------------------------------------------------------------------
__global__ __launch_bounds__(256)
void wtrans_kernel(const float* __restrict__ W, _Float16* __restrict__ BT,
                   int K, int N)
{
    __shared__ _Float16 t[32][33];
    const int kb = blockIdx.x * 32, nb = blockIdx.y * 32;
    const int tx = threadIdx.x, ty = threadIdx.y;   // 32 x 8
#pragma unroll
    for (int r = 0; r < 32; r += 8) {
        const int k = kb + ty + r, n = nb + tx;
        t[ty + r][tx] = (n < N) ? (_Float16)W[(size_t)k * N + n] : (_Float16)0.0f;
    }
    __syncthreads();
#pragma unroll
    for (int r = 0; r < 32; r += 8) {
        const int n = nb + ty + r, k = kb + tx;
        if (n < N) BT[(size_t)n * K + k] = t[tx][ty + r];
    }
}

// ---------------------------------------------------------------------------
// LayerNorm over last dim F (block per row), f32 in -> f16 out. act=1: SiLU.
// ---------------------------------------------------------------------------
__global__ __launch_bounds__(256)
void ln_kernel(const float* __restrict__ in, const float* __restrict__ g,
               const float* __restrict__ b, _Float16* __restrict__ out,
               int F, int act)
{
    const int row = blockIdx.x;
    const float* x = in + (size_t)row * F;
    _Float16* y = out + (size_t)row * F;

    float s = 0.0f, ss = 0.0f;
    for (int i = threadIdx.x; i < F; i += 256) {
        const float v = x[i];
        s += v; ss += v * v;
    }
#pragma unroll
    for (int o = 16; o > 0; o >>= 1) {
        s  += __shfl_xor(s, o, 32);
        ss += __shfl_xor(ss, o, 32);
    }
    __shared__ float shs[8], shss[8], mv[2];
    const int wv = threadIdx.x >> 5;
    if ((threadIdx.x & 31) == 0) { shs[wv] = s; shss[wv] = ss; }
    __syncthreads();
    if (threadIdx.x == 0) {
        float ts = 0.0f, tss = 0.0f;
        for (int i = 0; i < 8; ++i) { ts += shs[i]; tss += shss[i]; }
        const float m = ts / (float)F;
        mv[0] = m;
        mv[1] = rsqrtf(tss / (float)F - m * m + 1e-5f);
    }
    __syncthreads();
    const float m = mv[0], rinv = mv[1];
    for (int i = threadIdx.x; i < F; i += 256) {
        float v = (x[i] - m) * rinv * g[i] + b[i];
        if (act) v = v / (1.0f + __expf(-v));   // SiLU
        y[i] = (_Float16)v;
    }
}

// cls[b][c] = H[b][0][c]  (f16)
__global__ void gather_cls(const _Float16* __restrict__ Hs, _Float16* __restrict__ cls)
{
    const int i = blockIdx.x * 256 + threadIdx.x;   // over B*C
    const int b = i / CB, c = i % CB;
    cls[i] = Hs[(size_t)b * NTOK * CB + c];
}

// gv16[i] = (f16)( g[i] * silu(v[i]) )
__global__ void gv_kernel(const float* __restrict__ g, const float* __restrict__ v,
                          _Float16* __restrict__ out, size_t n)
{
    const size_t i = (size_t)blockIdx.x * 256 + threadIdx.x;
    if (i < n) {
        const float vv = v[i];
        out[i] = (_Float16)(g[i] * (vv / (1.0f + __expf(-vv))));
    }
}

// ---------------------------------------------------------------------------
// Fused attention: one block per (b, h). q/k LN + scale, scores + learned
// bias, stable softmax, PV. Tiny (65x65x64) -> VALU with padded LDS. f16 out.
// ---------------------------------------------------------------------------
__global__ __launch_bounds__(128)
void attn_kernel(const float* __restrict__ qkv, const float* __restrict__ bias,
                 const float* __restrict__ qg, const float* __restrict__ qb,
                 const float* __restrict__ kg, const float* __restrict__ kb,
                 _Float16* __restrict__ O)
{
    __shared__ float qs[NTOK][DHB + 1];
    __shared__ float ks[NTOK][DHB + 1];
    __shared__ float vs[NTOK][DHB + 1];
    __shared__ float pw[4][72];

    const int bh = blockIdx.x;
    const int b = bh / HB, h = bh % HB;
    const int wv = threadIdx.x >> 5, lane = threadIdx.x & 31;
    const float scale = 0.125f;   // DH^-0.5

    for (int n = wv; n < NTOK; n += 4) {
        const float* qp = qkv + ((size_t)(b * NTOK + n) * (3 * CB)) + h * DHB;
        {
            float v0 = qp[lane], v1 = qp[lane + 32];
            float s = v0 + v1, ss = v0 * v0 + v1 * v1;
#pragma unroll
            for (int o = 16; o > 0; o >>= 1) { s += __shfl_xor(s, o, 32); ss += __shfl_xor(ss, o, 32); }
            const float m = s * (1.0f / DHB);
            const float r = rsqrtf(ss * (1.0f / DHB) - m * m + 1e-5f);
            qs[n][lane]      = ((v0 - m) * r * qg[lane]      + qb[lane])      * scale;
            qs[n][lane + 32] = ((v1 - m) * r * qg[lane + 32] + qb[lane + 32]) * scale;
        }
        {
            const float* kp = qp + CB;
            float v0 = kp[lane], v1 = kp[lane + 32];
            float s = v0 + v1, ss = v0 * v0 + v1 * v1;
#pragma unroll
            for (int o = 16; o > 0; o >>= 1) { s += __shfl_xor(s, o, 32); ss += __shfl_xor(ss, o, 32); }
            const float m = s * (1.0f / DHB);
            const float r = rsqrtf(ss * (1.0f / DHB) - m * m + 1e-5f);
            ks[n][lane]      = (v0 - m) * r * kg[lane]      + kb[lane];
            ks[n][lane + 32] = (v1 - m) * r * kg[lane + 32] + kb[lane + 32];
        }
        {
            const float* vp = qp + 2 * CB;
            vs[n][lane]      = vp[lane];
            vs[n][lane + 32] = vp[lane + 32];
        }
    }
    __syncthreads();

    const float* bp = bias + (size_t)bh * (NTOK * NTOK);
    for (int n = wv; n < NTOK; n += 4) {
        float sc[3];
        const int mm[3] = { lane, lane + 32, lane + 64 };
        float mx = -1e30f;
#pragma unroll
        for (int t = 0; t < 3; ++t) {
            sc[t] = -1e30f;
            if (mm[t] < NTOK) {
                float d = 0.0f;
                for (int dd = 0; dd < DHB; ++dd) d += qs[n][dd] * ks[mm[t]][dd];
                d += bp[n * NTOK + mm[t]];
                sc[t] = d;
                mx = fmaxf(mx, d);
            }
        }
#pragma unroll
        for (int o = 16; o > 0; o >>= 1) mx = fmaxf(mx, __shfl_xor(mx, o, 32));
        float sum = 0.0f;
#pragma unroll
        for (int t = 0; t < 3; ++t) {
            if (mm[t] < NTOK) {
                const float e = __expf(sc[t] - mx);
                pw[wv][mm[t]] = e;
                sum += e;
            }
        }
#pragma unroll
        for (int o = 16; o > 0; o >>= 1) sum += __shfl_xor(sum, o, 32);
        const float inv = 1.0f / sum;
        __builtin_amdgcn_wave_barrier();   // same-wave LDS ordering fence
        for (int dd = lane; dd < DHB; dd += 32) {
            float a = 0.0f;
            for (int m2 = 0; m2 < NTOK; ++m2) a += pw[wv][m2] * vs[m2][dd];
            O[(size_t)(b * NTOK + n) * CB + h * DHB + dd] = (_Float16)(a * inv);
        }
        __builtin_amdgcn_wave_barrier();
    }
}

// ---------------------------------------------------------------------------
static void launch_gemm(const _Float16* A, int lda, const _Float16* BT, int ldb,
                        float* C, int ldc, int M, int N, int K, int accum,
                        hipStream_t s)
{
    dim3 grid((N + BN - 1) / BN, (M + BM - 1) / BM);
    gemm_f16wmma_async<<<grid, 256, 0, s>>>(A, lda, BT, ldb, C, ldc, M, N, K, accum);
}

static void launch_wtrans(const float* W, _Float16* BT, int K, int N, hipStream_t s)
{
    dim3 grid(K / 32, (N + 31) / 32);
    wtrans_kernel<<<grid, dim3(32, 8), 0, s>>>(W, BT, K, N);
}

extern "C" void kernel_launch(void* const* d_in, const int* in_sizes, int n_in,
                              void* d_out, int out_size, void* d_ws, size_t ws_size,
                              hipStream_t stream)
{
    (void)in_sizes; (void)n_in; (void)out_size; (void)ws_size;

    const float* x        = (const float*)d_in[0];
    const float* ln1_g    = (const float*)d_in[1];
    const float* ln1_b    = (const float*)d_in[2];
    const float* qkv_w    = (const float*)d_in[3];
    const float* proj_w   = (const float*)d_in[4];
    const float* qn_g     = (const float*)d_in[5];
    const float* qn_b     = (const float*)d_in[6];
    const float* kn_g     = (const float*)d_in[7];
    const float* kn_b     = (const float*)d_in[8];
    const float* sg_w1    = (const float*)d_in[9];
    const float* sg_ln1_g = (const float*)d_in[10];
    const float* sg_ln1_b = (const float*)d_in[11];
    const float* sg_w2    = (const float*)d_in[12];
    const float* sg_ln2_g = (const float*)d_in[13];
    const float* sg_ln2_b = (const float*)d_in[14];
    const float* sg_bw    = (const float*)d_in[15];
    const float* ln2_g    = (const float*)d_in[16];
    const float* ln2_b    = (const float*)d_in[17];
    const float* gate_w   = (const float*)d_in[18];
    const float* value_w  = (const float*)d_in[19];
    const float* out_w    = (const float*)d_in[20];

    float* X = (float*)d_out;    // resident activation, updated in place
    const size_t xElems = (size_t)BB * NTOK * CB;
    hipMemcpyAsync(X, x, xElems * sizeof(float), hipMemcpyDeviceToDevice, stream);

    // byte arena, 256B-aligned carve
    char* base = (char*)d_ws;
    size_t off = 0;
    auto carve = [&](size_t bytes) {
        char* p = base + off;
        off += (bytes + 255) & ~(size_t)255;
        return p;
    };
    _Float16* H16  = (_Float16*)carve(xElems * 2);                        // LN out (h / h2)
    float*    QKV  = (float*)   carve((size_t)BB * NTOK * 3 * CB * 4);
    _Float16* O16  = (_Float16*)carve(xElems * 2);                        // attn out
    float*    BIAS = (float*)   carve((size_t)BB * HB * NTOK * NTOK * 4);
    float*    G    = (float*)   carve((size_t)BB * NTOK * 4 * CB * 4);
    float*    Vv   = (float*)   carve((size_t)BB * NTOK * 4 * CB * 4);
    _Float16* GV16 = (_Float16*)carve((size_t)BB * NTOK * 4 * CB * 2);
    _Float16* CLS  = (_Float16*)carve((size_t)BB * CB * 2);
    float*    L1f  = (float*)   carve((size_t)BB * 2 * LATB * 4);
    _Float16* L1h  = (_Float16*)carve((size_t)BB * 2 * LATB * 2);
    float*    L2f  = (float*)   carve((size_t)BB * LATB * HB * 4);
    _Float16* L2h  = (_Float16*)carve((size_t)BB * LATB * HB * 2);
    _Float16* BT   = (_Float16*)carve((size_t)4 * CB * CB * 2);           // largest weight (f16, transposed)

    const int rowsBN = BB * NTOK;           // 33280, % 128 == 0
    const size_t nGV = (size_t)rowsBN * 4 * CB;

    for (int l = 0; l < 6; ++l) {
        const float* l1g = ln1_g + l * CB;
        const float* l1b = ln1_b + l * CB;
        const float* qkw = qkv_w + (size_t)l * CB * 3 * CB;
        const float* prw = proj_w + (size_t)l * CB * CB;
        const float* w1  = sg_w1 + (size_t)l * CB * 2 * LATB;
        const float* s1g = sg_ln1_g + l * 2 * LATB;
        const float* s1b = sg_ln1_b + l * 2 * LATB;
        const float* w2  = sg_w2 + (size_t)l * 2 * LATB * LATB * HB;
        const float* s2g = sg_ln2_g + l * LATB * HB;
        const float* s2b = sg_ln2_b + l * LATB * HB;
        const float* bw  = sg_bw + (size_t)l * LATB * NTOK * NTOK;
        const float* l2g = ln2_g + l * CB;
        const float* l2b = ln2_b + l * CB;
        const float* gw  = gate_w + (size_t)l * CB * 4 * CB;
        const float* vw  = value_w + (size_t)l * CB * 4 * CB;
        const float* ow  = out_w + (size_t)l * 4 * CB * CB;

        // 1. h = LN1(x) -> f16
        ln_kernel<<<rowsBN, 256, 0, stream>>>(X, l1g, l1b, H16, CB, 0);
        // 2. qkv = h @ qkv_w
        launch_wtrans(qkw, BT, CB, 3 * CB, stream);
        launch_gemm(H16, CB, BT, CB, QKV, 3 * CB, rowsBN, 3 * CB, CB, 0, stream);
        // 3. bias path from cls token
        gather_cls<<<(BB * CB) / 256, 256, 0, stream>>>(H16, CLS);
        launch_wtrans(w1, BT, CB, 2 * LATB, stream);
        launch_gemm(CLS, CB, BT, CB, L1f, 2 * LATB, BB, 2 * LATB, CB, 0, stream);
        ln_kernel<<<BB, 256, 0, stream>>>(L1f, s1g, s1b, L1h, 2 * LATB, 1);   // LN + SiLU
        launch_wtrans(w2, BT, 2 * LATB, LATB * HB, stream);
        launch_gemm(L1h, 2 * LATB, BT, 2 * LATB, L2f, LATB * HB, BB, LATB * HB, 2 * LATB, 0, stream);
        ln_kernel<<<BB, 256, 0, stream>>>(L2f, s2g, s2b, L2h, LATB * HB, 0);
        //    bias = lat2(B*H, 256) @ bw(256, 4225)
        launch_wtrans(bw, BT, LATB, NTOK * NTOK, stream);
        launch_gemm(L2h, LATB, BT, LATB, BIAS, NTOK * NTOK,
                    BB * HB, NTOK * NTOK, LATB, 0, stream);
        // 4. attention -> f16 O
        attn_kernel<<<BB * HB, 128, 0, stream>>>(QKV, BIAS,
                                                 qn_g + l * DHB, qn_b + l * DHB,
                                                 kn_g + l * DHB, kn_b + l * DHB, O16);
        // 5. x += o @ proj_w
        launch_wtrans(prw, BT, CB, CB, stream);
        launch_gemm(O16, CB, BT, CB, X, CB, rowsBN, CB, CB, 1, stream);
        // 6. h2 = LN2(x) -> f16
        ln_kernel<<<rowsBN, 256, 0, stream>>>(X, l2g, l2b, H16, CB, 0);
        // 7/8. gate & value
        launch_wtrans(gw, BT, CB, 4 * CB, stream);
        launch_gemm(H16, CB, BT, CB, G, 4 * CB, rowsBN, 4 * CB, CB, 0, stream);
        launch_wtrans(vw, BT, CB, 4 * CB, stream);
        launch_gemm(H16, CB, BT, CB, Vv, 4 * CB, rowsBN, 4 * CB, CB, 0, stream);
        // 9. GV = G * silu(V) -> f16
        gv_kernel<<<(unsigned)((nGV + 255) / 256), 256, 0, stream>>>(G, Vv, GV16, nGV);
        // 10. x += GV @ out_w
        launch_wtrans(ow, BT, 4 * CB, CB, stream);
        launch_gemm(GV16, 4 * CB, BT, 4 * CB, X, CB, rowsBN, CB, 4 * CB, 1, stream);
    }
}
